// CausalSelfAttention_36189394436533
// MI455X (gfx1250) — matrix operand order repaired and checked by reading, not compiled, not run on hardware
//
#include <hip/hip_runtime.h>
#include <hip/hip_bf16.h>

// ---------------------------------------------------------------------------
// CausalSelfAttention (S=4096, D=1024, H=16, hd=64) for gfx1250 (MI455X).
// All matmuls on v_wmma_f32_16x16x32_bf16 (bf16 operands, fp32 accum).
// GEMMs use 32x64 per-wave tiles with all fragments staged into distinct
// registers per k-step (lets the scheduler use partial s_wait_loadcnt).
// Flash attention overlaps V-block fetch (async-to-LDS when available) with
// the softmax VALU/exp section.
// ---------------------------------------------------------------------------

#define S_LEN   4096
#define D_MODEL 1024
#define N_HEADS 16
#define HEAD_D  64
#define ATT_SCALE 0.125f                    // 1/sqrt(64)
#define LOG2_ROPE_BASE 13.287712379549449f  // log2(10000)

#if __has_builtin(__builtin_amdgcn_global_load_async_to_lds_b128)
#define HAS_ASYNC_LDS 1
#else
#define HAS_ASYNC_LDS 0
#endif

typedef __attribute__((ext_vector_type(16))) __bf16 v16bf;
typedef __attribute__((ext_vector_type(8)))  float  v8f;

// Types matching the async-to-LDS builtin's parameters:
//   param0: int __attribute__((vector_size(16))) __device__ *   (addrspace 1)
//   param1: same vector type in LDS                              (addrspace 3)
typedef int v4i_raw __attribute__((vector_size(16)));
typedef __attribute__((address_space(1))) v4i_raw* global_v4i_ptr;
typedef __attribute__((address_space(3))) v4i_raw* lds_v4i_ptr;

union BF16Frag {
    v16bf v;
    unsigned int u[8];
    unsigned short us[16];
};

// fp32 -> bf16, round to nearest even
static __device__ __forceinline__ unsigned short f2bf(float f) {
    union { float f; unsigned u; } x; x.f = f;
    unsigned r = x.u + 0x7FFFu + ((x.u >> 16) & 1u);
    return (unsigned short)(r >> 16);
}
static __device__ __forceinline__ float bf2f(unsigned short b) {
    union { unsigned u; float f; } x; x.u = ((unsigned)b) << 16;
    return x.f;
}

// 16-byte load of 8 bf16 into 4 dwords of a fragment
static __device__ __forceinline__ void load8bf(unsigned int* dst, const unsigned short* p) {
    const uint4* q = reinterpret_cast<const uint4*>(p);
    uint4 t = *q;
    dst[0] = t.x; dst[1] = t.y; dst[2] = t.z; dst[3] = t.w;
}

static __device__ __forceinline__ v8f wmma_bf16(const v16bf& a, const v16bf& b, v8f c) {
    return __builtin_amdgcn_wmma_f32_16x16x32_bf16(false, a, false, b, (short)0, c,
                                                   false, false);
}

static __device__ __forceinline__ void wait_async0() {
#if __has_builtin(__builtin_amdgcn_s_wait_asynccnt)
    __builtin_amdgcn_s_wait_asynccnt(0);
#else
    asm volatile("s_wait_asynccnt 0" ::: "memory");
#endif
}

#if HAS_ASYNC_LDS
// One 16-byte async global->LDS copy per active lane (tracked by ASYNCcnt).
static __device__ __forceinline__ void async_copy_b128(const unsigned short* gp,
                                                       unsigned short* lp) {
    __builtin_amdgcn_global_load_async_to_lds_b128(
        (global_v4i_ptr)(uintptr_t)gp,
        (lds_v4i_ptr)(unsigned int)(uintptr_t)lp,
        0, 0);
}
#endif

// ---------------------------------------------------------------------------
// Kernel 0: fp32 -> bf16 conversion
// ---------------------------------------------------------------------------
__global__ void cvt_f32_to_bf16(const float* __restrict__ src,
                                unsigned short* __restrict__ dst, int n) {
    int i = blockIdx.x * blockDim.x + threadIdx.x;
    if (i < n) dst[i] = f2bf(src[i]);
}

// ---------------------------------------------------------------------------
// Kernel 1: Q/K/V = x @ W^T.  One wave = 32(M) x 64(N) tile, k step 32.
//   A-frag (16x32): lane row = base + lane%16; k groups at 8*(lane/16) and +16.
//   B-frag (32x16): lane col = lane%16 of W row; k groups at
//   8*(lane/16) and +16 (same as A).
//   All 12 B128 loads of a k-step staged before the 8 WMMAs.
// ---------------------------------------------------------------------------
__global__ __launch_bounds__(128)
void qkv_gemm(const unsigned short* __restrict__ xb,
              const unsigned short* __restrict__ Wqb,
              const unsigned short* __restrict__ Wkb,
              const unsigned short* __restrict__ Wvb,
              unsigned short* __restrict__ Qb,
              unsigned short* __restrict__ Kb,
              unsigned short* __restrict__ Vtb) {
    int gw   = blockIdx.x * (blockDim.x >> 5) + (threadIdx.x >> 5);
    int lane = threadIdx.x & 31;
    int mat  = gw / (128 * 16);      // 0=Q 1=K 2=V
    int rem  = gw % (128 * 16);
    int sb   = rem >> 4;             // 32-row block of sequence (0..127)
    int nt   = rem & 15;             // 64-col tile of output dim
    const unsigned short* W = (mat == 0) ? Wqb : (mat == 1) ? Wkb : Wvb;

    int half = lane >> 4;
    int l16  = lane & 15;
    const unsigned short* arow0 = xb + (size_t)(sb * 32 + l16) * D_MODEL;
    const unsigned short* arow1 = arow0 + (size_t)16 * D_MODEL;

    v8f z = {0.f,0.f,0.f,0.f,0.f,0.f,0.f,0.f};
    v8f acc[2][4];
#pragma unroll
    for (int m = 0; m < 2; m++)
        for (int t = 0; t < 4; t++) acc[m][t] = z;

    for (int k = 0; k < D_MODEL; k += 32) {
        __builtin_prefetch(arow0 + k + 64, 0, 0);
        __builtin_prefetch(arow1 + k + 64, 0, 0);
        BF16Frag a0, a1, b[4];
        load8bf(a0.u,     arow0 + k + half * 8);
        load8bf(a0.u + 4, arow0 + k + half * 8 + 16);
        load8bf(a1.u,     arow1 + k + half * 8);
        load8bf(a1.u + 4, arow1 + k + half * 8 + 16);
#pragma unroll
        for (int t = 0; t < 4; t++) {
            const unsigned short* wrow =
                W + (size_t)(nt * 64 + t * 16 + l16) * D_MODEL + k + half * 8;
            load8bf(b[t].u,     wrow);
            load8bf(b[t].u + 4, wrow + 16);
        }
#pragma unroll
        for (int t = 0; t < 4; t++) {
            acc[0][t] = wmma_bf16(a0.v, b[t].v, acc[0][t]);
            acc[1][t] = wmma_bf16(a1.v, b[t].v, acc[1][t]);
        }
    }

#pragma unroll
    for (int m = 0; m < 2; m++) {
#pragma unroll
        for (int t = 0; t < 4; t++) {
            int ocol = nt * 64 + t * 16 + l16;
            int h = ocol >> 6, d = ocol & 63;
#pragma unroll
            for (int r = 0; r < 8; r++) {
                int s = sb * 32 + m * 16 + r + half * 8;
                unsigned short bv = f2bf(acc[m][t][r]);
                if (mat == 0)      Qb[((size_t)h * S_LEN + s) * HEAD_D + d] = bv;
                else if (mat == 1) Kb[((size_t)h * S_LEN + s) * HEAD_D + d] = bv;
                else               Vtb[((size_t)h * HEAD_D + d) * S_LEN + s] = bv;
            }
        }
    }
}

// ---------------------------------------------------------------------------
// Kernel 2: RoPE in place on Q and K.  Thread owns pair (j, j+32), j in [0,32).
// ---------------------------------------------------------------------------
__global__ void rope_kernel(unsigned short* __restrict__ Qb,
                            unsigned short* __restrict__ Kb) {
    int i = blockIdx.x * blockDim.x + threadIdx.x;   // H*S*32 threads
    if (i >= N_HEADS * S_LEN * 32) return;
    int j = i & 31;
    int s = (i >> 5) & (S_LEN - 1);
    int h = i >> 17;
    float inv = exp2f(-((float)j / 32.0f) * LOG2_ROPE_BASE);
    float ang = (float)s * inv;
    float c = __cosf(ang), sn = __sinf(ang);
    size_t base = ((size_t)h * S_LEN + s) * HEAD_D;

    float q0 = bf2f(Qb[base + j]), q1 = bf2f(Qb[base + j + 32]);
    Qb[base + j]      = f2bf(q0 * c - q1 * sn);
    Qb[base + j + 32] = f2bf(q1 * c + q0 * sn);

    float k0 = bf2f(Kb[base + j]), k1 = bf2f(Kb[base + j + 32]);
    Kb[base + j]      = f2bf(k0 * c - k1 * sn);
    Kb[base + j + 32] = f2bf(k1 * c + k0 * sn);
}

// ---------------------------------------------------------------------------
// Kernel 3: flash attention.  One wave = 16 queries of one head; key blocks of
// 32; online softmax; probs bounced through LDS to rebuild the A-fragment.
// V-block fetch overlaps the softmax section: async copy to LDS when the
// gfx1250 builtin is available, otherwise staged direct loads.
// Blocks group 4 heads at the same q-block so __syncthreads() is convergent.
// ---------------------------------------------------------------------------
__global__ __launch_bounds__(128)
void flash_attn(const unsigned short* __restrict__ Qb,
                const unsigned short* __restrict__ Kb,
                const unsigned short* __restrict__ Vtb,
                unsigned short* __restrict__ Ob) {
    __shared__ unsigned short Pbuf[4][16 * 32];
#if HAS_ASYNC_LDS
    __shared__ unsigned short Vbuf[4][64 * 32];   // 64 d-rows x 32 keys per wave
#endif

    int wid  = threadIdx.x >> 5;
    int lane = threadIdx.x & 31;
    int qb   = blockIdx.x & 255;          // same q-block for all 4 waves
    int h    = (blockIdx.x >> 8) * 4 + wid;
    int qbase = qb * 16;
    int half = lane >> 4, l16 = lane & 15;

    // Q A-fragments for d in [0,32) and [32,64)
    BF16Frag qa[2];
    const unsigned short* qrow = Qb + ((size_t)h * S_LEN + qbase + l16) * HEAD_D;
#pragma unroll
    for (int g = 0; g < 2; g++) {
        load8bf(qa[g].u,     qrow + g * 32 + half * 16);
        load8bf(qa[g].u + 4, qrow + g * 32 + half * 16 + 8);
    }

    float m_i[8], l_i[8];
    v8f o[4];
    v8f z = {0.f,0.f,0.f,0.f,0.f,0.f,0.f,0.f};
#pragma unroll
    for (int r = 0; r < 8; r++) { m_i[r] = -1e30f; l_i[r] = 0.f; }
#pragma unroll
    for (int t = 0; t < 4; t++) o[t] = z;

    unsigned short* P = Pbuf[wid];
    const unsigned short* vbase = Vtb + (size_t)h * HEAD_D * S_LEN;
    int nkb = (qbase + 15) / 32 + 1;      // uniform across the block

    for (int kb = 0; kb < nkb; kb++) {
        bool maskblk = (kb * 32 + 31 > qbase);

#if HAS_ASYNC_LDS
        // ---- kick off async V-block copy (64 d-rows x 32 keys = 4 KB).
        // 8 instructions; each lane moves 16B; lane -> (row, quarter).
        {
            unsigned short* Vw = Vbuf[wid];
            int rq  = lane >> 2;          // row within group of 8
            int qtr = lane & 3;           // 16B quarter of the 64B row
#pragma unroll
            for (int g = 0; g < 8; g++) {
                int row = g * 8 + rq;
                const unsigned short* gp =
                    vbase + (size_t)row * S_LEN + kb * 32 + qtr * 8;
                unsigned short* lp = Vw + row * 32 + qtr * 8;
                async_copy_b128(gp, lp);
            }
        }
#endif

        // ---- K fragments staged (8 B128 into distinct regs), then 4 WMMAs
        BF16Frag bk[2][2];
#pragma unroll
        for (int t = 0; t < 2; t++) {
            const unsigned short* krow =
                Kb + ((size_t)h * S_LEN + kb * 32 + t * 16 + l16) * HEAD_D;
#pragma unroll
            for (int g = 0; g < 2; g++) {
                load8bf(bk[t][g].u,     krow + g * 32 + half * 16);
                load8bf(bk[t][g].u + 4, krow + g * 32 + half * 16 + 8);
            }
        }
        v8f sacc[2]; sacc[0] = z; sacc[1] = z;
#pragma unroll
        for (int t = 0; t < 2; t++)
#pragma unroll
            for (int g = 0; g < 2; g++)
                sacc[t] = wmma_bf16(qa[g].v, bk[t][g].v, sacc[t]);

#if !HAS_ASYNC_LDS
        // ---- direct V loads issued now so they are in flight during softmax
        BF16Frag bv[4];
#pragma unroll
        for (int t = 0; t < 4; t++) {
            const unsigned short* vp =
                vbase + (size_t)(t * 16 + l16) * S_LEN + kb * 32 + half * 16;
            load8bf(bv[t].u,     vp);
            load8bf(bv[t].u + 4, vp + 8);
        }
#endif

        // ---- online softmax (rows live in 16-lane halves; butterfly reduce)
        float pr0[8], pr1[8];
#pragma unroll
        for (int r = 0; r < 8; r++) {
            int q = qbase + r + half * 8;
            float s0 = sacc[0][r] * ATT_SCALE;
            float s1 = sacc[1][r] * ATT_SCALE;
            if (maskblk) {
                if (kb * 32 + l16 > q)      s0 = -1e30f;
                if (kb * 32 + 16 + l16 > q) s1 = -1e30f;
            }
            float rmax = fmaxf(s0, s1);
            for (int d = 8; d >= 1; d >>= 1)
                rmax = fmaxf(rmax, __shfl_xor(rmax, d, 32));
            float mnew = fmaxf(m_i[r], rmax);
            float corr = __expf(m_i[r] - mnew);
            m_i[r] = mnew;
            float p0 = __expf(s0 - mnew);
            float p1 = __expf(s1 - mnew);
            float rsum = p0 + p1;
            for (int d = 8; d >= 1; d >>= 1)
                rsum += __shfl_xor(rsum, d, 32);
            l_i[r] = l_i[r] * corr + rsum;
#pragma unroll
            for (int t = 0; t < 4; t++) o[t][r] *= corr;
            pr0[r] = p0; pr1[r] = p1;
        }

        // ---- C-layout probs -> 16x32 bf16 A-fragment via LDS
#pragma unroll
        for (int r = 0; r < 8; r++) {
            int m = r + half * 8;
            P[m * 32 + l16]      = f2bf(pr0[r]);
            P[m * 32 + 16 + l16] = f2bf(pr1[r]);
        }
        __syncthreads();

        BF16Frag pa;
        {
            const unsigned short* prow = P + l16 * 32 + half * 16;
            const uint4* p0 = reinterpret_cast<const uint4*>(prow);
            const uint4* p1 = reinterpret_cast<const uint4*>(prow + 8);
            uint4 t0 = *p0, t1 = *p1;
            pa.u[0] = t0.x; pa.u[1] = t0.y; pa.u[2] = t0.z; pa.u[3] = t0.w;
            pa.u[4] = t1.x; pa.u[5] = t1.y; pa.u[6] = t1.z; pa.u[7] = t1.w;
        }

#if HAS_ASYNC_LDS
        // ---- consume async-staged V block from LDS
        wait_async0();
        BF16Frag bv[4];
        {
            const unsigned short* Vw = Vbuf[wid];
#pragma unroll
            for (int t = 0; t < 4; t++) {
                const unsigned short* vr = Vw + (t * 16 + l16) * 32 + half * 16;
                const uint4* v0 = reinterpret_cast<const uint4*>(vr);
                const uint4* v1 = reinterpret_cast<const uint4*>(vr + 8);
                uint4 t0 = *v0, t1 = *v1;
                bv[t].u[0] = t0.x; bv[t].u[1] = t0.y; bv[t].u[2] = t0.z; bv[t].u[3] = t0.w;
                bv[t].u[4] = t1.x; bv[t].u[5] = t1.y; bv[t].u[6] = t1.z; bv[t].u[7] = t1.w;
            }
        }
#endif

        // ---- O += P @ V
#pragma unroll
        for (int t = 0; t < 4; t++)
            o[t] = wmma_bf16(pa.v, bv[t].v, o[t]);
        __syncthreads();
    }

    // ---- normalize and store to Ob (S, D) bf16
#pragma unroll
    for (int t = 0; t < 4; t++) {
        int dcol = t * 16 + l16;
#pragma unroll
        for (int r = 0; r < 8; r++) {
            int s = qbase + r + half * 8;
            Ob[(size_t)s * D_MODEL + h * HEAD_D + dcol] = f2bf(o[t][r] / l_i[r]);
        }
    }
}

// ---------------------------------------------------------------------------
// Kernel 4: out = Ob @ Wo^T  (fp32 output, 32x64 per-wave tiles)
// ---------------------------------------------------------------------------
__global__ __launch_bounds__(128)
void out_gemm(const unsigned short* __restrict__ Ob,
              const unsigned short* __restrict__ Wob,
              float* __restrict__ out) {
    int gw   = blockIdx.x * (blockDim.x >> 5) + (threadIdx.x >> 5);
    int lane = threadIdx.x & 31;
    int sb = gw >> 4;                 // 0..127
    int nt = gw & 15;
    int half = lane >> 4, l16 = lane & 15;
    const unsigned short* arow0 = Ob + (size_t)(sb * 32 + l16) * D_MODEL;
    const unsigned short* arow1 = arow0 + (size_t)16 * D_MODEL;

    v8f z = {0.f,0.f,0.f,0.f,0.f,0.f,0.f,0.f};
    v8f acc[2][4];
#pragma unroll
    for (int m = 0; m < 2; m++)
        for (int t = 0; t < 4; t++) acc[m][t] = z;

    for (int k = 0; k < D_MODEL; k += 32) {
        __builtin_prefetch(arow0 + k + 64, 0, 0);
        __builtin_prefetch(arow1 + k + 64, 0, 0);
        BF16Frag a0, a1, b[4];
        load8bf(a0.u,     arow0 + k + half * 8);
        load8bf(a0.u + 4, arow0 + k + half * 8 + 16);
        load8bf(a1.u,     arow1 + k + half * 8);
        load8bf(a1.u + 4, arow1 + k + half * 8 + 16);
#pragma unroll
        for (int t = 0; t < 4; t++) {
            const unsigned short* wrow =
                Wob + (size_t)(nt * 64 + t * 16 + l16) * D_MODEL + k + half * 8;
            load8bf(b[t].u,     wrow);
            load8bf(b[t].u + 4, wrow + 16);
        }
#pragma unroll
        for (int t = 0; t < 4; t++) {
            acc[0][t] = wmma_bf16(a0.v, b[t].v, acc[0][t]);
            acc[1][t] = wmma_bf16(a1.v, b[t].v, acc[1][t]);
        }
    }

#pragma unroll
    for (int m = 0; m < 2; m++) {
#pragma unroll
        for (int t = 0; t < 4; t++) {
            int ocol = nt * 64 + t * 16 + l16;
#pragma unroll
            for (int r = 0; r < 8; r++) {
                int s = sb * 32 + m * 16 + r + half * 8;
                out[(size_t)s * D_MODEL + ocol] = acc[m][t][r];
            }
        }
    }
}

// ---------------------------------------------------------------------------
// Host-side launcher
// ---------------------------------------------------------------------------
extern "C" void kernel_launch(void* const* d_in, const int* in_sizes, int n_in,
                              void* d_out, int out_size, void* d_ws, size_t ws_size,
                              hipStream_t stream) {
    const float* x  = (const float*)d_in[0];
    const float* Wq = (const float*)d_in[1];
    const float* Wk = (const float*)d_in[2];
    const float* Wv = (const float*)d_in[3];
    const float* Wo = (const float*)d_in[4];
    float* out = (float*)d_out;

    const size_t MB = 1024 * 1024;
    char* ws = (char*)d_ws;
    unsigned short* xb  = (unsigned short*)(ws + 0 * MB);   // 8 MB
    unsigned short* Wqb = (unsigned short*)(ws + 8 * MB);   // 2 MB
    unsigned short* Wkb = (unsigned short*)(ws + 10 * MB);  // 2 MB
    unsigned short* Wvb = (unsigned short*)(ws + 12 * MB);  // 2 MB
    unsigned short* Wob = (unsigned short*)(ws + 14 * MB);  // 2 MB
    unsigned short* Qb  = (unsigned short*)(ws + 16 * MB);  // 8 MB
    unsigned short* Kb  = (unsigned short*)(ws + 24 * MB);  // 8 MB
    unsigned short* Vtb = (unsigned short*)(ws + 32 * MB);  // 8 MB
    unsigned short* Ob  = (unsigned short*)(ws + 40 * MB);  // 8 MB

    const int nX = S_LEN * D_MODEL;
    const int nW = D_MODEL * D_MODEL;
    cvt_f32_to_bf16<<<(nX + 255) / 256, 256, 0, stream>>>(x,  xb,  nX);
    cvt_f32_to_bf16<<<(nW + 255) / 256, 256, 0, stream>>>(Wq, Wqb, nW);
    cvt_f32_to_bf16<<<(nW + 255) / 256, 256, 0, stream>>>(Wk, Wkb, nW);
    cvt_f32_to_bf16<<<(nW + 255) / 256, 256, 0, stream>>>(Wv, Wvb, nW);
    cvt_f32_to_bf16<<<(nW + 255) / 256, 256, 0, stream>>>(Wo, Wob, nW);

    // 3 mats * 128 s-blocks * 16 n-tiles = 6144 waves, 4 waves/block
    qkv_gemm<<<1536, 128, 0, stream>>>(xb, Wqb, Wkb, Wvb, Qb, Kb, Vtb);

    rope_kernel<<<(N_HEADS * S_LEN * 32 + 255) / 256, 256, 0, stream>>>(Qb, Kb);

    // 256 q-blocks * 4 head-groups; 4 heads (waves) per block
    flash_attn<<<1024, 128, 0, stream>>>(Qb, Kb, Vtb, Ob);

    // 128 s-blocks * 16 n-tiles = 2048 waves, 4 waves/block
    out_gemm<<<512, 128, 0, stream>>>(Ob, Wob, out);
}
